// IntentClassifier_27625229648090
// MI455X (gfx1250) — matrix operand order repaired and checked by reading, not compiled, not run on hardware
//
#include <hip/hip_runtime.h>
#include <cmath>

// ---------------------------------------------------------------------------
// MI455X / gfx1250, wave32. Matrix cores via v_wmma_f32_16x16x32_bf16,
// LSTM state staging via Tensor Data Mover (tensor_load_to_lds + TENSORcnt).
// ---------------------------------------------------------------------------

typedef __attribute__((ext_vector_type(16))) __bf16 v16bf;
typedef __attribute__((ext_vector_type(8)))  __bf16 v8bf;
typedef __attribute__((ext_vector_type(8)))  float  v8f;
typedef unsigned int u32x4 __attribute__((ext_vector_type(4)));
typedef int          i32x8 __attribute__((ext_vector_type(8)));
typedef int          i32x4 __attribute__((ext_vector_type(4)));

union BF16F { v16bf v; v8bf h[2]; };

__device__ __forceinline__ v8f wmma_bf16(const v16bf& a, const v16bf& b, const v8f& c){
    // D = A(16x32 bf16) x B(32x16 bf16) + C(16x16 f32)
    return __builtin_amdgcn_wmma_f32_16x16x32_bf16(false, a, false, b, (short)0, c, false, false);
}

__device__ __forceinline__ float sigf(float x){ return 1.0f / (1.0f + __expf(-x)); }

#define TT   512
#define HH   768
#define G4H  3072
#define E2H  1536

// ---------------------------------------------------------------------------
// TDM: async-load a 2D bf16 tile (32 elems wide x 64 rows, tensor row stride
// 768 elems) from global into LDS with row padding 64B data + 16B pad, which
// reproduces the padded __shared__ [64][40] layout.  D# per ISA 08 §8.3/8.4.
// amdgpu-toolchain (clang-23) 6-arg builtin form.
// ---------------------------------------------------------------------------
__device__ __forceinline__ void tdm_load_h_chunk(unsigned lds_off, const __bf16* gsrc)
{
    const unsigned long long ga = (unsigned long long)(size_t)gsrc;
    u32x4 g0;
    g0[0] = 1u;                                                 // count=1 (valid user D#)
    g0[1] = lds_off;                                            // LDS byte address
    g0[2] = (unsigned)ga;                                       // global_addr[95:64]
    g0[3] = (unsigned)((ga >> 32) & 0x01FFFFFFu) | (2u << 30);  // addr[120:96] | type=2
    i32x8 g1;
    g1[0] = (int)((1u << 16) |      // data_size = 1 -> 2 bytes
                  (1u << 20) |      // pad_enable
                  (3u << 22) |      // pad_interval code 3 -> every 16 DWORDs (64B)
                  (3u << 25));      // pad_amount  code 3 -> 4 DWORDs (16B)
    g1[1] = (int)(768u << 16);                  // tensor_dim0 = 768 (lo16)
    g1[2] = (int)((768u >> 16) | (64u << 16));  // tensor_dim0 hi | tensor_dim1 = 64
    g1[3] = (int)(32u << 16);                   // tensor_dim1 hi | tile_dim0 = 32
    g1[4] = (int)(64u);                         // tile_dim1 = 64, tile_dim2 = 0
    g1[5] = (int)768;                           // tensor_dim0_stride lo32 (elems)
    g1[6] = 0;                                  // stride0 hi | stride1 lo
    g1[7] = 0;                                  // stride1 hi
    const i32x4 gz4 = {0, 0, 0, 0};             // groups 2/3 unused (2D tensor)
    const i32x8 gz8 = {0, 0, 0, 0, 0, 0, 0, 0};
    __builtin_amdgcn_tensor_load_to_lds(g0, g1, gz4, gz4, gz8, 0);
}

// ---------------------------------------------------------------------------
// Generic tiled GEMM:  C[M,N] = act(A[M,K] @ W[K,N] + bias)
// fp32 in/out, bf16 WMMA compute.  Block tile 64x128, 8 waves (2x4), K-step 32.
// Requires M%64==0, N%128==0, K%32==0.
// ---------------------------------------------------------------------------
__global__ __launch_bounds__(256)
void gemm_bf16_kernel(const float* __restrict__ A, const float* __restrict__ W,
                      const float* __restrict__ bias, float* __restrict__ C,
                      int M, int K, int N, int act)
{
    __shared__ __bf16 aS[64][40];    // [m][k], +8 pad vs LDS bank conflicts
    __shared__ __bf16 bS[128][40];   // [n][k] (transposed so K is contiguous)
    const int tid = threadIdx.x;
    const int lid = tid & 31, w = tid >> 5;
    const int wm = w >> 2, wn = w & 3;               // wave -> 32x32 sub-tile
    const int m0 = blockIdx.y * 64, n0 = blockIdx.x * 128;
    v8f acc[2][2] = {};

    for (int k0 = 0; k0 < K; k0 += 32){
        __syncthreads();
        {   // stage A 64x32 (convert f32 -> bf16)
            const int r = tid >> 2, cs = (tid & 3) * 8;
            const float* src = A + (size_t)(m0 + r) * K + k0 + cs;
            #pragma unroll
            for (int j = 0; j < 8; ++j) aS[r][cs + j] = (__bf16)src[j];
        }
        // stage B 32x128 transposed -> bS[n][k]
        #pragma unroll
        for (int j = 0; j < 16; ++j){
            const int idx = tid + 256 * j;
            const int kk = idx >> 7, c = idx & 127;
            bS[c][kk] = (__bf16)W[(size_t)(k0 + kk) * N + n0 + c];
        }
        __syncthreads();

        #pragma unroll
        for (int nt = 0; nt < 2; ++nt){
            // B frag (32x16): lane holds col=lid%16, K = (e<8?0:16)+8*(lid/16)+(e%8)
            BF16F bf;
            const int bc = 32*wn + nt*16 + (lid & 15);
            const int kh = 8 * (lid >> 4);
            bf.h[0] = *(const v8bf*)&bS[bc][kh];
            bf.h[1] = *(const v8bf*)&bS[bc][kh + 16];
            #pragma unroll
            for (int mt = 0; mt < 2; ++mt){
                // A frag (16x32): lane row=lid%16, K = (e<8?0:16)+8*(lid/16)+(e%8)
                BF16F af;
                const int ar = 32*wm + mt*16 + (lid & 15);
                const int h8 = 8 * (lid >> 4);
                af.h[0] = *(const v8bf*)&aS[ar][h8];
                af.h[1] = *(const v8bf*)&aS[ar][16 + h8];
                acc[mt][nt] = wmma_bf16(af.v, bf.v, acc[mt][nt]);
            }
        }
    }
    // epilogue: D layout row = r + 8*(lid/16), col = lid%16
    #pragma unroll
    for (int mt = 0; mt < 2; ++mt)
    #pragma unroll
    for (int nt = 0; nt < 2; ++nt){
        const int col = n0 + 32*wn + nt*16 + (lid & 15);
        const float bv = bias ? bias[col] : 0.0f;
        #pragma unroll
        for (int r = 0; r < 8; ++r){
            const int row = m0 + 32*wm + mt*16 + r + 8*(lid >> 4);
            float v = acc[mt][nt][r] + bv;
            if (act == 1) v = tanhf(v);
            C[(size_t)row * N + col] = v;
        }
    }
}

// ---------------------------------------------------------------------------
// Wh [768,3072] f32  ->  WhT [3072,768] bf16  (K contiguous for B-fragments)
// ---------------------------------------------------------------------------
__global__ void convert_whT_kernel(const float* __restrict__ Wh, __bf16* __restrict__ WhT){
    const int idx = blockIdx.x * 256 + threadIdx.x;   // n*768 + k
    const int n = idx / HH, k = idx - n * HH;
    WhT[idx] = (__bf16)Wh[(size_t)k * G4H + n];
}

// ---------------------------------------------------------------------------
// device-wide barrier (24 blocks, monotone counter, memset to 0 per launch)
// ---------------------------------------------------------------------------
__device__ __forceinline__ void grid_sync(unsigned* cnt, unsigned target){
    __threadfence();
    __syncthreads();
    if (threadIdx.x == 0){
        __hip_atomic_fetch_add(cnt, 1u, __ATOMIC_RELEASE, __HIP_MEMORY_SCOPE_AGENT);
        while (__hip_atomic_load(cnt, __ATOMIC_ACQUIRE, __HIP_MEMORY_SCOPE_AGENT) < target)
            __builtin_amdgcn_s_sleep(2);
    }
    __syncthreads();
}

// ---------------------------------------------------------------------------
// LSTM recurrence, one direction.  24 persistent WGs; WG wg owns hidden cols
// [32*wg, 32*wg+32) and therefore all four gate columns of that slice.
// Wh slice (768x128 bf16, transposed) lives in LDS across all 512 steps;
// per step only h (96KB, L2-resident) is re-staged, chunk-by-chunk via the
// Tensor Data Mover, double-buffered against the WMMA pipeline.
// xp already contains x@Wi + b.  dir=1 processes p=(s<L)?L-1-s:s (flip-free).
// ---------------------------------------------------------------------------
__global__ __launch_bounds__(256)
void lstm_rec_kernel(const float* __restrict__ xp,     // [64][512][3072]
                     const __bf16* __restrict__ whT,   // [3072][768]
                     const int*   __restrict__ lens,   // [64]
                     float*       __restrict__ enc,    // [64][512][1536]
                     __bf16*      __restrict__ h_bf,   // [64][768] shared state
                     unsigned*    __restrict__ gcnt, int dir)
{
    __shared__ __bf16 whS[128][776];                       // B slice [localcol][k]
    __shared__ __attribute__((aligned(16))) __bf16 hS[2][64][40];  // TDM double buffer
    __shared__ float  zS[64][132];                         // z slice [64][4*32]
    __shared__ float  cS[64][32];                          // cell state (LDS-resident)
    __shared__ int    lenS[64];

    const int tid = threadIdx.x, lid = tid & 31, w = tid >> 5;
    const int wg = blockIdx.x;        // 0..23

    // ---- preload Wh slice: local col c -> global z-col (c/32)*768 + 32*wg + (c%32)
    #pragma unroll 4
    for (int it = 0; it < 48; ++it){
        const int g8 = tid + 256 * it;            // group of 8 bf16
        const int c  = g8 / 96, gk = (g8 % 96) * 8;
        const int gcol = (c >> 5) * HH + 32 * wg + (c & 31);
        *(v8bf*)&whS[c][gk] = *(const v8bf*)&whT[(size_t)gcol * HH + gk];
    }
    if (tid < 64) lenS[tid] = lens[tid];
    // ---- zero h / c state for this WG's slice
    for (int i = tid; i < 64 * 32; i += 256){
        const int b = i >> 5, jj = i & 31;
        cS[b][jj] = 0.0f;
        h_bf[b * HH + 32 * wg + jj] = (__bf16)0.0f;
    }

    const unsigned hs_off[2] = { (unsigned)(size_t)&hS[0][0][0],
                                 (unsigned)(size_t)&hS[1][0][0] };
    unsigned phase = 0;
    const int bcol = w * 16 + (lid & 15);   // local B column 0..127
    const int bkh  = 8 * (lid >> 4);
    const int ah8  = 8  * (lid >> 4);
    const int am   = lid & 15;

    for (int s = 0; s < TT; ++s){
        grid_sync(gcnt, 24u * (++phase));   // all h_bf writes of step s-1 visible

        v8f acc[4] = {};
        if (w == 0) tdm_load_h_chunk(hs_off[0], h_bf);        // chunk 0
        for (int kc = 0; kc < 24; ++kc){    // K = 768 = 24 x 32
            const int kb = kc * 32;
            if (w == 0) __builtin_amdgcn_s_wait_tensorcnt(0); // chunk kc landed
            __syncthreads();                                  // visible to all waves
            if (w == 0 && kc < 23)                            // prefetch next chunk
                tdm_load_h_chunk(hs_off[(kc + 1) & 1], h_bf + kb + 32);
            const int buf = kc & 1;
            BF16F bf;
            bf.h[0] = *(const v8bf*)&whS[bcol][kb + bkh];
            bf.h[1] = *(const v8bf*)&whS[bcol][kb + bkh + 16];
            #pragma unroll
            for (int mt = 0; mt < 4; ++mt){ // M = 64 = 4 x 16 batch rows
                BF16F af;
                af.h[0] = *(const v8bf*)&hS[buf][mt*16 + am][ah8];
                af.h[1] = *(const v8bf*)&hS[buf][mt*16 + am][16 + ah8];
                acc[mt] = wmma_bf16(af.v, bf.v, acc[mt]);
            }
        }
        // write z tile to LDS (D layout: row = mt*16 + r + 8*(lid/16), col = lid%16)
        #pragma unroll
        for (int mt = 0; mt < 4; ++mt)
            #pragma unroll
            for (int r = 0; r < 8; ++r)
                zS[mt*16 + r + ah8][w*16 + am] = acc[mt][r];

        // everyone must finish reading h_bf for step s before gate phase rewrites it
        grid_sync(gcnt, 24u * (++phase));

        // ---- gate phase: element (b, jj), 8 per thread
        #pragma unroll
        for (int e = 0; e < 8; ++e){
            const int flat = tid + 256 * e;
            const int b = flat >> 5, jj = flat & 31;
            const int Lb = lenS[b];
            int p = s;
            if (dir) p = (s < Lb) ? (Lb - 1 - s) : s;
            const size_t xrow  = (size_t)b * TT + p;
            const size_t xbase = xrow * (size_t)G4H + 32 * wg + jj;
            const float zi = zS[b][jj]      + xp[xbase];
            const float zf = zS[b][32 + jj] + xp[xbase + 768];
            const float zg = zS[b][64 + jj] + xp[xbase + 1536];
            const float zo = zS[b][96 + jj] + xp[xbase + 2304];
            const float co = cS[b][jj];
            const float nc = sigf(zf) * co + sigf(zi) * tanhf(zg);
            const float nh = sigf(zo) * tanhf(nc);
            enc[xrow * (size_t)E2H + dir * HH + 32 * wg + jj] = nh;
            if (s < Lb){                      // masked state update
                cS[b][jj] = nc;
                h_bf[b * HH + 32 * wg + jj] = (__bf16)nh;
            }
        }
    }
}

// ---------------------------------------------------------------------------
// scores[b,t] = sum_n energy[b,t,n] * We[n];  masked to -inf for t >= len[b]
// ---------------------------------------------------------------------------
__global__ __launch_bounds__(256)
void scores_kernel(const float* __restrict__ energy, const float* __restrict__ we,
                   const int* __restrict__ lens, float* __restrict__ scores){
    const int row = blockIdx.x * 8 + (threadIdx.x >> 5);   // one wave per (b,t)
    const int lid = threadIdx.x & 31;
    const float* e = energy + (size_t)row * E2H;
    float s = 0.0f;
    for (int j = lid; j < E2H; j += 32) s += e[j] * we[j];
    #pragma unroll
    for (int off = 16; off > 0; off >>= 1) s += __shfl_down(s, off, 32);
    if (lid == 0){
        const int b = row >> 9, t = row & 511;
        scores[row] = (t < lens[b]) ? s : -INFINITY;
    }
}

__global__ __launch_bounds__(256)
void softmax_kernel(float* __restrict__ scores){
    __shared__ float red[256];
    const int b = blockIdx.x, tid = threadIdx.x;
    float* s = scores + (size_t)b * TT;
    float m = -INFINITY;
    for (int t = tid; t < TT; t += 256) m = fmaxf(m, s[t]);
    red[tid] = m; __syncthreads();
    for (int st = 128; st > 0; st >>= 1){ if (tid < st) red[tid] = fmaxf(red[tid], red[tid+st]); __syncthreads(); }
    m = red[0]; __syncthreads();
    float ev[2]; float loc = 0.0f;
    { int i = 0; for (int t = tid; t < TT; t += 256, ++i){ ev[i] = __expf(s[t] - m); loc += ev[i]; } }
    red[tid] = loc; __syncthreads();
    for (int st = 128; st > 0; st >>= 1){ if (tid < st) red[tid] += red[tid+st]; __syncthreads(); }
    const float inv = 1.0f / red[0];
    { int i = 0; for (int t = tid; t < TT; t += 256, ++i) s[t] = ev[i] * inv; }
}

__global__ __launch_bounds__(256)
void context_kernel(const float* __restrict__ attn, const float* __restrict__ enc,
                    float* __restrict__ ctx){
    const int b = blockIdx.x, tid = threadIdx.x;
    float acc[6] = {0,0,0,0,0,0};
    const float* e = enc + (size_t)b * TT * E2H;
    const float* a = attn + (size_t)b * TT;
    for (int t = 0; t < TT; ++t){
        const float wv = a[t];
        const float* er = e + (size_t)t * E2H;
        #pragma unroll
        for (int j = 0; j < 6; ++j) acc[j] += wv * er[tid + 256 * j];
    }
    #pragma unroll
    for (int j = 0; j < 6; ++j) ctx[(size_t)b * E2H + tid + 256 * j] = acc[j];
}

__global__ void intents_kernel(const float* __restrict__ ctx, const float* __restrict__ Wint,
                               const float* __restrict__ bint, float* __restrict__ out){
    const int b = blockIdx.x, i = threadIdx.x;
    if (i < 60){
        float s = bint[i];
        const float* c = ctx + (size_t)b * E2H;
        for (int n = 0; n < E2H; ++n) s += c[n] * Wint[n * 60 + i];
        out[b * 60 + i] = s;
    }
}

// ---------------------------------------------------------------------------
extern "C" void kernel_launch(void* const* d_in, const int* in_sizes, int n_in,
                              void* d_out, int out_size, void* d_ws, size_t ws_size,
                              hipStream_t stream)
{
    (void)in_sizes; (void)n_in; (void)out_size; (void)ws_size;
    const float* lattice = (const float*)d_in[0];
    const int*   lens    = (const int*)d_in[1];
    const float* W_reg = (const float*)d_in[3];
    const float* b_reg = (const float*)d_in[4];
    const float* Wi0f  = (const float*)d_in[5];
    const float* Wh0f  = (const float*)d_in[6];
    const float* b0f   = (const float*)d_in[7];
    const float* Wi0b  = (const float*)d_in[8];
    const float* Wh0b  = (const float*)d_in[9];
    const float* b0b   = (const float*)d_in[10];
    const float* Wi1f  = (const float*)d_in[11];
    const float* Wh1f  = (const float*)d_in[12];
    const float* b1f   = (const float*)d_in[13];
    const float* Wi1b  = (const float*)d_in[14];
    const float* Wh1b  = (const float*)d_in[15];
    const float* b1b   = (const float*)d_in[16];
    const float* W_keys   = (const float*)d_in[17];
    const float* W_energy = (const float*)d_in[18];
    const float* W_post   = (const float*)d_in[19];
    const float* b_post   = (const float*)d_in[20];
    const float* W_int    = (const float*)d_in[21];
    const float* b_int    = (const float*)d_in[22];

    char* ws = (char*)d_ws;
    float*  reg    = (float*) (ws + 0ull);                 // 128 MB
    float*  xp     = (float*) (ws + 134217728ull);         // 384 MB (reused as energy)
    float*  enc0   = (float*) (ws + 536870912ull);         // 192 MB
    float*  enc1   = (float*) (ws + 738197504ull);         // 192 MB
    __bf16* whT0f  = (__bf16*)(ws + 939524096ull);
    __bf16* whT0b  = (__bf16*)(ws + 944242688ull);
    __bf16* whT1f  = (__bf16*)(ws + 948961280ull);
    __bf16* whT1b  = (__bf16*)(ws + 953679872ull);
    __bf16* h_bf   = (__bf16*)(ws + 958398464ull);
    float*  scores = (float*) (ws + 958496768ull);
    float*  ctx    = (float*) (ws + 958627840ull);
    float*  ctxp   = (float*) (ws + 959021056ull);
    unsigned* gcnt = (unsigned*)(ws + 959414272ull);
    float*  energy = xp;

    const int M = 64 * TT;   // 32768

    // bf16-transpose the recurrent weights once
    convert_whT_kernel<<<9216, 256, 0, stream>>>(Wh0f, whT0f);
    convert_whT_kernel<<<9216, 256, 0, stream>>>(Wh0b, whT0b);
    convert_whT_kernel<<<9216, 256, 0, stream>>>(Wh1f, whT1f);
    convert_whT_kernel<<<9216, 256, 0, stream>>>(Wh1b, whT1b);

    // reg = lattice @ W_reg + b_reg
    gemm_bf16_kernel<<<dim3(8,  M/64), 256, 0, stream>>>(lattice, W_reg, b_reg, reg, M, 1024, 1024, 0);

    // ---- layer 0 forward / backward
    gemm_bf16_kernel<<<dim3(24, M/64), 256, 0, stream>>>(reg, Wi0f, b0f, xp, M, 1024, G4H, 0);
    (void)hipMemsetAsync(gcnt, 0, sizeof(unsigned), stream);
    lstm_rec_kernel<<<24, 256, 0, stream>>>(xp, whT0f, lens, enc0, h_bf, gcnt, 0);

    gemm_bf16_kernel<<<dim3(24, M/64), 256, 0, stream>>>(reg, Wi0b, b0b, xp, M, 1024, G4H, 0);
    (void)hipMemsetAsync(gcnt, 0, sizeof(unsigned), stream);
    lstm_rec_kernel<<<24, 256, 0, stream>>>(xp, whT0b, lens, enc0, h_bf, gcnt, 1);

    // ---- layer 1 forward / backward
    gemm_bf16_kernel<<<dim3(24, M/64), 256, 0, stream>>>(enc0, Wi1f, b1f, xp, M, E2H, G4H, 0);
    (void)hipMemsetAsync(gcnt, 0, sizeof(unsigned), stream);
    lstm_rec_kernel<<<24, 256, 0, stream>>>(xp, whT1f, lens, enc1, h_bf, gcnt, 0);

    gemm_bf16_kernel<<<dim3(24, M/64), 256, 0, stream>>>(enc0, Wi1b, b1b, xp, M, E2H, G4H, 0);
    (void)hipMemsetAsync(gcnt, 0, sizeof(unsigned), stream);
    lstm_rec_kernel<<<24, 256, 0, stream>>>(xp, whT1b, lens, enc1, h_bf, gcnt, 1);

    // ---- attention + head
    gemm_bf16_kernel<<<dim3(12, M/64), 256, 0, stream>>>(enc1, W_keys, nullptr, energy, M, E2H, E2H, 1);
    scores_kernel <<<M/8, 256, 0, stream>>>(energy, W_energy, lens, scores);
    softmax_kernel<<<64,  256, 0, stream>>>(scores);
    context_kernel<<<64,  256, 0, stream>>>(scores, enc1, ctx);
    gemm_bf16_kernel<<<dim3(12, 1), 256, 0, stream>>>(ctx, W_post, b_post, ctxp, 64, E2H, E2H, 0);
    intents_kernel<<<64, 64, 0, stream>>>(ctxp, W_int, b_int, (float*)d_out);
}
